// CrossAttentionBlock_19335942767092
// MI455X (gfx1250) — matrix operand order repaired and checked by reading, not compiled, not run on hardware
//
#include <hip/hip_runtime.h>
#include <hip/hip_bf16.h>

// ---------------------------------------------------------------------------
// CrossAttentionBlock for MI455X (gfx1250, wave32, WMMA bf16 path)
// B=16, C=512, H=W=64 (HW=4096), T=77, TEXT_DIM=768, HEADS=8, HEAD_DIM=64
// ---------------------------------------------------------------------------

#define DEVFN static __device__ __forceinline__

typedef __attribute__((ext_vector_type(8)))  __bf16 bf16x8;
typedef __attribute__((ext_vector_type(16))) __bf16 bf16x16;
typedef __attribute__((ext_vector_type(8)))  float  f32x8;

DEVFN __bf16 f2bf(float f) {
  unsigned u = __builtin_bit_cast(unsigned, f);
  unsigned r = u + 0x7FFFu + ((u >> 16) & 1u);   // round-to-nearest-even
  unsigned short h = (unsigned short)(r >> 16);
  return __builtin_bit_cast(__bf16, h);
}

DEVFN __bf16 bf_zero() { return __builtin_bit_cast(__bf16, (unsigned short)0); }

DEVFN f32x8 zero8() {
  f32x8 z;
#pragma unroll
  for (int i = 0; i < 8; ++i) z[i] = 0.0f;
  return z;
}

DEVFN bf16x16 zerobf() {
  bf16x16 z;
#pragma unroll
  for (int i = 0; i < 16; ++i) z[i] = bf_zero();
  return z;
}

DEVFN bf16x16 combine8(bf16x8 lo, bf16x8 hi) {
  bf16x16 r;
#pragma unroll
  for (int i = 0; i < 8; ++i) { r[i] = lo[i]; r[8 + i] = hi[i]; }
  return r;
}

// A-matrix fragment (16x32 bf16, MxK). Per ISA: lane<16 -> K {0..7,16..23},
// lane>=16 -> K {8..15,24..31}. row = per-lane row pointer (M = lane&15).
DEVFN bf16x16 load_afrag(const __bf16* row, int ck, int lane) {
  int koff = (lane & 16) ? 8 : 0;
  bf16x8 lo = *(const bf16x8*)(row + ck + koff);
  bf16x8 hi = *(const bf16x8*)(row + ck + 16 + koff);
  return combine8(lo, hi);
}

// B-matrix fragment (32x16 bf16, KxN). lane<16 -> K {0..7,16..23}, lane>=16 -> K {8..15,24..31}
// (same order as A). row = per-lane row pointer indexed by N (= lane&15),
// memory holds W[N][K] with K contiguous.
DEVFN bf16x16 load_bfrag(const __bf16* row, int ck, int lane) {
  int koff = (lane & 16) ? 8 : 0;
  bf16x8 lo = *(const bf16x8*)(row + ck + koff);
  bf16x8 hi = *(const bf16x8*)(row + ck + 16 + koff);
  return combine8(lo, hi);
}

DEVFN f32x8 wmma_bf16(bf16x16 a, bf16x16 b, f32x8 c) {
  return __builtin_amdgcn_wmma_f32_16x16x32_bf16(false, a, false, b,
                                                 (short)0, c, false, false);
}

// ---------------------------------------------------------------------------
// Kernel 1: fp32 -> bf16 convert
// ---------------------------------------------------------------------------
__global__ void cvt_f32_bf16(const float* __restrict__ src,
                             __bf16* __restrict__ dst, int n) {
  int i = blockIdx.x * 256 + threadIdx.x;
  if (i < n) dst[i] = f2bf(src[i]);
}

// ---------------------------------------------------------------------------
// Kernel 2: GroupNorm statistics. One block per (b,g); 16 ch x 4096 px each.
// ---------------------------------------------------------------------------
__global__ void gn_stats_k(const float* __restrict__ x, float* __restrict__ stats) {
  int bg = blockIdx.x;                        // b*32 + g
  const float4* base = (const float4*)(x + (size_t)bg * 16 * 4096);
  float s = 0.f, ss = 0.f;
  for (int i = threadIdx.x; i < 16384; i += 256) {
    float4 v = base[i];
    s  += v.x + v.y + v.z + v.w;
    ss += v.x * v.x + v.y * v.y + v.z * v.z + v.w * v.w;
  }
  __shared__ float rs[256], rss[256];
  rs[threadIdx.x] = s; rss[threadIdx.x] = ss;
  __syncthreads();
  for (int st = 128; st > 0; st >>= 1) {
    if (threadIdx.x < st) {
      rs[threadIdx.x]  += rs[threadIdx.x + st];
      rss[threadIdx.x] += rss[threadIdx.x + st];
    }
    __syncthreads();
  }
  if (threadIdx.x == 0) {
    float mean = rs[0] * (1.0f / 65536.0f);
    float var  = rss[0] * (1.0f / 65536.0f) - mean * mean;
    stats[bg * 2]     = mean;
    stats[bg * 2 + 1] = rsqrtf(var + 1e-5f);
  }
}

// ---------------------------------------------------------------------------
// Kernel 3: apply GN, transpose x[b][c][s] -> xnT[b][s][c] (bf16)
// ---------------------------------------------------------------------------
__global__ void gn_apply_k(const float* __restrict__ x,
                           const float* __restrict__ stats,
                           const float* __restrict__ gamma,
                           const float* __restrict__ beta,
                           __bf16* __restrict__ xnT) {
  int b = blockIdx.z, cb = blockIdx.y * 64, sb = blockIdx.x * 64;
  __shared__ float tile[64][65];
  int tid = threadIdx.x;
  int sl = tid & 63, c0 = tid >> 6;
#pragma unroll 4
  for (int i = 0; i < 16; ++i) {
    int cl = c0 + i * 4;
    int c  = cb + cl;
    int g  = b * 32 + (c >> 4);
    float mu = stats[g * 2], rstd = stats[g * 2 + 1];
    float v = x[((size_t)(b * 512 + c)) * 4096 + sb + sl];
    tile[cl][sl] = (v - mu) * rstd * gamma[c] + beta[c];
  }
  __syncthreads();
  int cl2 = tid & 63, s0 = tid >> 6;
#pragma unroll 4
  for (int i = 0; i < 16; ++i) {
    int sl2 = s0 + i * 4;
    xnT[((size_t)b * 4096 + sb + sl2) * 512 + cb + cl2] = f2bf(tile[cl2][sl2]);
  }
}

// ---------------------------------------------------------------------------
// Kernel 4: Q GEMM  q[s,o] = sum_c xnT[s,c]*qw[o,c] + qb[o]
// output as qT[b][h][s][d] bf16 (d contiguous for attention A-fragments)
// block = 8 waves, block tile 128(M=s) x 64(N=o), wave tile 32x32, K=512
// ---------------------------------------------------------------------------
__global__ void q_gemm_k(const __bf16* __restrict__ xnT,
                         const __bf16* __restrict__ qw,
                         const float* __restrict__ qb,
                         __bf16* __restrict__ qT) {
  int b = blockIdx.z;
  int m0 = blockIdx.y * 128;
  int n0 = blockIdx.x * 64;
  int wave = threadIdx.x >> 5, lane = threadIdx.x & 31;
  int ms = m0 + (wave >> 1) * 32;
  int ns = n0 + (wave & 1) * 32;
  int la = lane & 15;

  const __bf16* Abase = xnT + (size_t)b * 4096 * 512;
  const __bf16* arow0 = Abase + (size_t)(ms + la) * 512;
  const __bf16* arow1 = Abase + (size_t)(ms + 16 + la) * 512;
  const __bf16* brow0 = qw + (size_t)(ns + la) * 512;
  const __bf16* brow1 = qw + (size_t)(ns + 16 + la) * 512;

  f32x8 acc[2][2];
#pragma unroll
  for (int i = 0; i < 2; ++i)
#pragma unroll
    for (int j = 0; j < 2; ++j) acc[i][j] = zero8();

  for (int k = 0; k < 512; k += 32) {
    if (k + 64 < 512) {
      __builtin_prefetch(arow0 + k + 64, 0, 0);
      __builtin_prefetch(arow1 + k + 64, 0, 0);
    }
    bf16x16 a0 = load_afrag(arow0, k, lane);
    bf16x16 a1 = load_afrag(arow1, k, lane);
    bf16x16 b0 = load_bfrag(brow0, k, lane);
    bf16x16 b1 = load_bfrag(brow1, k, lane);
    acc[0][0] = wmma_bf16(a0, b0, acc[0][0]);
    acc[0][1] = wmma_bf16(a0, b1, acc[0][1]);
    acc[1][0] = wmma_bf16(a1, b0, acc[1][0]);
    acc[1][1] = wmma_bf16(a1, b1, acc[1][1]);
  }

  int crow = (lane >> 4) * 8, ccol = lane & 15;
#pragma unroll
  for (int mt = 0; mt < 2; ++mt)
#pragma unroll
    for (int nt = 0; nt < 2; ++nt) {
      int o = ns + nt * 16 + ccol;
      float bias = qb[o];
      int h = o >> 6, d = o & 63;
#pragma unroll
      for (int r = 0; r < 8; ++r) {
        int s = ms + mt * 16 + crow + r;
        qT[(((size_t)(b * 8 + h) * 4096) + s) * 64 + d] =
            f2bf(acc[mt][nt][r] + bias);
      }
    }
}

// ---------------------------------------------------------------------------
// Kernel 5: K/V linear. klin[b][t][o] = sum_f text[b,t,f]*w[o,f] + bias[o]
// M=t (tiles of 16, pad 77->80), N=512 (8 waves x 64), K=768
// ---------------------------------------------------------------------------
__global__ void kv_gemm_k(const __bf16* __restrict__ text,
                          const __bf16* __restrict__ kw,
                          const __bf16* __restrict__ vw,
                          const float* __restrict__ kb,
                          const float* __restrict__ vb,
                          __bf16* __restrict__ klin,
                          __bf16* __restrict__ vlin) {
  int b = blockIdx.z;
  int which = blockIdx.y;                 // 0=K, 1=V
  int t0 = blockIdx.x * 16;
  const __bf16* w    = which ? vw : kw;
  const float*  bias = which ? vb : kb;
  __bf16*       out  = which ? vlin : klin;

  int wave = threadIdx.x >> 5, lane = threadIdx.x & 31;
  int n0 = wave * 64;
  int la = lane & 15;
  int trow = t0 + la; if (trow > 76) trow = 76;   // clamp OOB rows for loads
  const __bf16* arow = text + ((size_t)b * 77 + trow) * 768;
  const __bf16* brow[4];
#pragma unroll
  for (int nt = 0; nt < 4; ++nt)
    brow[nt] = w + (size_t)(n0 + nt * 16 + la) * 768;

  f32x8 acc[4];
#pragma unroll
  for (int nt = 0; nt < 4; ++nt) acc[nt] = zero8();

  for (int k = 0; k < 768; k += 32) {
    bf16x16 a = load_afrag(arow, k, lane);
#pragma unroll
    for (int nt = 0; nt < 4; ++nt)
      acc[nt] = wmma_bf16(a, load_bfrag(brow[nt], k, lane), acc[nt]);
  }

  int crow = (lane >> 4) * 8, ccol = lane & 15;
#pragma unroll
  for (int nt = 0; nt < 4; ++nt) {
    int o = n0 + nt * 16 + ccol;
    float bv = bias[o];
#pragma unroll
    for (int r = 0; r < 8; ++r) {
      int t = t0 + crow + r;
      if (t < 77)
        out[((size_t)b * 77 + t) * 512 + o] = f2bf(acc[nt][r] + bv);
    }
  }
}

// ---------------------------------------------------------------------------
// Kernel 6: attention per (b, head, chunk of 256 query rows).
// klin viewed as [b][h][d][t] (flat index h*4928 + d*77 + t  == torch view).
// S = (Q K)*scale, masked softmax over t (77 -> pad 80), O = P V^T.
// ---------------------------------------------------------------------------
__global__ void attn_k(const __bf16* __restrict__ qT,
                       const __bf16* __restrict__ klin,
                       const __bf16* __restrict__ vlin,
                       __bf16* __restrict__ oT) {
  int b = blockIdx.z, h = blockIdx.y, chunk = blockIdx.x;

  __shared__ __align__(16) unsigned short ldsK[80 * 64];       // [t][d]
  __shared__ __align__(16) unsigned short ldsV[64 * 80];       // [d][t]
  __shared__ __align__(16) unsigned short ldsP[8 * 32 * 80];   // per-wave [s][t]

  const __bf16* kb_ = klin + (size_t)b * 39424 + h * 4928;
  const __bf16* vb_ = vlin + (size_t)b * 39424 + h * 4928;
  for (int idx = threadIdx.x; idx < 5120; idx += 256) {
    int d = idx / 80, t = idx % 80;
    unsigned short kv = (t < 77)
        ? __builtin_bit_cast(unsigned short, kb_[d * 77 + t]) : (unsigned short)0;
    unsigned short vv = (t < 77)
        ? __builtin_bit_cast(unsigned short, vb_[d * 77 + t]) : (unsigned short)0;
    ldsK[t * 64 + d] = kv;     // transposed for B-fragments of QK^T
    ldsV[d * 80 + t] = vv;     // t-contiguous for B-fragments of P*V^T
  }
  __syncthreads();

  int wave = threadIdx.x >> 5, lane = threadIdx.x & 31;
  int la = lane & 15;
  int koff8 = (lane & 16) ? 8 : 0;
  int srow0 = chunk * 256 + wave * 32;

  // --- Q fragments (A: 16x32, K = head_dim, 2 K-steps, 2 M-tiles) ---
  const __bf16* qbase = qT + (size_t)(b * 8 + h) * 4096 * 64;
  bf16x16 aq[2][2];
#pragma unroll
  for (int mt = 0; mt < 2; ++mt) {
    const __bf16* row = qbase + (size_t)(srow0 + mt * 16 + la) * 64;
    aq[mt][0] = load_afrag(row, 0, lane);
    aq[mt][1] = load_afrag(row, 32, lane);
  }

  // --- S = Q K : 2 M-tiles x 5 N-tiles (t), K=64 ---
  f32x8 S[2][5];
#pragma unroll
  for (int mt = 0; mt < 2; ++mt)
#pragma unroll
    for (int nt = 0; nt < 5; ++nt) S[mt][nt] = zero8();

#pragma unroll
  for (int nt = 0; nt < 5; ++nt) {
    const __bf16* brow = (const __bf16*)ldsK + (nt * 16 + la) * 64;
    bf16x16 b0 = load_bfrag(brow, 0, lane);
    bf16x16 b1 = load_bfrag(brow, 32, lane);
#pragma unroll
    for (int mt = 0; mt < 2; ++mt) {
      S[mt][nt] = wmma_bf16(aq[mt][0], b0, S[mt][nt]);
      S[mt][nt] = wmma_bf16(aq[mt][1], b1, S[mt][nt]);
    }
  }

  // --- masked softmax over t (per output row) ---
  const float scale = 0.125f;       // 1/sqrt(64)
  int col = lane & 15;
  int rowoff = (lane >> 4) * 8;
#pragma unroll
  for (int mt = 0; mt < 2; ++mt) {
    float p[5][8];
#pragma unroll
    for (int nt = 0; nt < 5; ++nt)
#pragma unroll
      for (int r = 0; r < 8; ++r) {
        int t = nt * 16 + col;
        float v = S[mt][nt][r] * scale;
        p[nt][r] = (t < 77) ? v : -3.0e38f;
      }
#pragma unroll
    for (int r = 0; r < 8; ++r) {
      float m = p[0][r];
#pragma unroll
      for (int nt = 1; nt < 5; ++nt) m = fmaxf(m, p[nt][r]);
      m = fmaxf(m, __shfl_xor(m, 1, 32));
      m = fmaxf(m, __shfl_xor(m, 2, 32));
      m = fmaxf(m, __shfl_xor(m, 4, 32));
      m = fmaxf(m, __shfl_xor(m, 8, 32));
      float sum = 0.f;
#pragma unroll
      for (int nt = 0; nt < 5; ++nt) {
        float e = __expf(p[nt][r] - m);
        p[nt][r] = e;
        sum += e;
      }
      sum += __shfl_xor(sum, 1, 32);
      sum += __shfl_xor(sum, 2, 32);
      sum += __shfl_xor(sum, 4, 32);
      sum += __shfl_xor(sum, 8, 32);
      float inv = 1.0f / sum;
#pragma unroll
      for (int nt = 0; nt < 5; ++nt) p[nt][r] *= inv;
    }
    unsigned short* pw = ldsP + wave * 32 * 80 + mt * 16 * 80;
#pragma unroll
    for (int nt = 0; nt < 5; ++nt)
#pragma unroll
      for (int r = 0; r < 8; ++r)
        pw[(rowoff + r) * 80 + nt * 16 + col] =
            __builtin_bit_cast(unsigned short, f2bf(p[nt][r]));
  }
  __syncthreads();

  // --- O = P * V^T : 2 M-tiles x 4 N-tiles (d), K = t (80, 2.5 steps) ---
  f32x8 accO[2][4];
#pragma unroll
  for (int mt = 0; mt < 2; ++mt)
#pragma unroll
    for (int nd = 0; nd < 4; ++nd) accO[mt][nd] = zero8();

  for (int ks = 0; ks < 3; ++ks) {
    bf16x16 bv[4];
#pragma unroll
    for (int nd = 0; nd < 4; ++nd) {
      const __bf16* brow = (const __bf16*)ldsV + (nd * 16 + la) * 80;
      if (ks < 2) {
        bv[nd] = load_bfrag(brow, ks * 32, lane);
      } else {
        bf16x16 z = zerobf();
        {                                      // K = 64..79 valid; 80..95 zero
          bf16x8 lo = *(const bf16x8*)(brow + 64 + koff8);
          for (int i = 0; i < 8; ++i) z[i] = lo[i];
        }
        bv[nd] = z;
      }
    }
#pragma unroll
    for (int mt = 0; mt < 2; ++mt) {
      const __bf16* arow =
          (const __bf16*)ldsP + (size_t)(wave * 32 + mt * 16 + la) * 80;
      bf16x16 ap;
      if (ks < 2) {
        ap = load_afrag(arow, ks * 32, lane);
      } else {
        bf16x16 z = zerobf();
        bf16x8 lo = *(const bf16x8*)(arow + 64 + koff8);   // t = 64..79
#pragma unroll
        for (int i = 0; i < 8; ++i) z[i] = lo[i];
        ap = z;
      }
#pragma unroll
      for (int nd = 0; nd < 4; ++nd)
        accO[mt][nd] = wmma_bf16(ap, bv[nd], accO[mt][nd]);
    }
  }

  // store oT[b][s][h*64+d]
  __bf16* ob = oT + (size_t)b * 4096 * 512 + h * 64;
#pragma unroll
  for (int mt = 0; mt < 2; ++mt)
#pragma unroll
    for (int nd = 0; nd < 4; ++nd)
#pragma unroll
      for (int r = 0; r < 8; ++r) {
        int s = srow0 + mt * 16 + rowoff + r;
        int d = nd * 16 + col;
        ob[(size_t)s * 512 + d] = f2bf(accO[mt][nd][r]);
      }
}

// ---------------------------------------------------------------------------
// Kernel 7: projection GEMM + bias + residual -> fp32 out[b][o][s]
// ---------------------------------------------------------------------------
__global__ void proj_gemm_k(const __bf16* __restrict__ oT,
                            const __bf16* __restrict__ pw,
                            const float* __restrict__ pb,
                            const float* __restrict__ x,
                            float* __restrict__ out) {
  int b = blockIdx.z;
  int m0 = blockIdx.y * 128;
  int n0 = blockIdx.x * 64;
  int wave = threadIdx.x >> 5, lane = threadIdx.x & 31;
  int ms = m0 + (wave >> 1) * 32;
  int ns = n0 + (wave & 1) * 32;
  int la = lane & 15;

  const __bf16* Abase = oT + (size_t)b * 4096 * 512;
  const __bf16* arow0 = Abase + (size_t)(ms + la) * 512;
  const __bf16* arow1 = Abase + (size_t)(ms + 16 + la) * 512;
  const __bf16* brow0 = pw + (size_t)(ns + la) * 512;
  const __bf16* brow1 = pw + (size_t)(ns + 16 + la) * 512;

  f32x8 acc[2][2];
#pragma unroll
  for (int i = 0; i < 2; ++i)
#pragma unroll
    for (int j = 0; j < 2; ++j) acc[i][j] = zero8();

  for (int k = 0; k < 512; k += 32) {
    if (k + 64 < 512) {
      __builtin_prefetch(arow0 + k + 64, 0, 0);
      __builtin_prefetch(arow1 + k + 64, 0, 0);
    }
    bf16x16 a0 = load_afrag(arow0, k, lane);
    bf16x16 a1 = load_afrag(arow1, k, lane);
    bf16x16 b0 = load_bfrag(brow0, k, lane);
    bf16x16 b1 = load_bfrag(brow1, k, lane);
    acc[0][0] = wmma_bf16(a0, b0, acc[0][0]);
    acc[0][1] = wmma_bf16(a0, b1, acc[0][1]);
    acc[1][0] = wmma_bf16(a1, b0, acc[1][0]);
    acc[1][1] = wmma_bf16(a1, b1, acc[1][1]);
  }

  int crow = (lane >> 4) * 8, ccol = lane & 15;
#pragma unroll
  for (int mt = 0; mt < 2; ++mt)
#pragma unroll
    for (int nt = 0; nt < 2; ++nt) {
      int o = ns + nt * 16 + ccol;
      float bias = pb[o];
#pragma unroll
      for (int r = 0; r < 8; ++r) {
        int s = ms + mt * 16 + crow + r;
        size_t gi = ((size_t)(b * 512 + o)) * 4096 + s;
        out[gi] = acc[mt][nt][r] + bias + x[gi];
      }
    }
}

// ---------------------------------------------------------------------------
// Host launcher
// ---------------------------------------------------------------------------
extern "C" void kernel_launch(void* const* d_in, const int* in_sizes, int n_in,
                              void* d_out, int out_size, void* d_ws, size_t ws_size,
                              hipStream_t stream) {
  const float* x     = (const float*)d_in[0];
  const float* text  = (const float*)d_in[1];
  const float* gamma = (const float*)d_in[2];
  const float* beta  = (const float*)d_in[3];
  const float* qw    = (const float*)d_in[4];
  const float* qb    = (const float*)d_in[5];
  const float* kw    = (const float*)d_in[6];
  const float* kb    = (const float*)d_in[7];
  const float* vw    = (const float*)d_in[8];
  const float* vb    = (const float*)d_in[9];
  const float* pw    = (const float*)d_in[10];
  const float* pb    = (const float*)d_in[11];
  float* out = (float*)d_out;

  char* ws = (char*)d_ws;
  size_t off = 0;
  auto carve = [&](size_t bytes) -> char* {
    char* p = ws + off;
    off = (off + bytes + 255) & ~(size_t)255;
    return p;
  };
  float*  stats = (float*) carve((size_t)512 * 2 * sizeof(float));
  __bf16* qw_bf = (__bf16*)carve((size_t)512 * 512 * 2);
  __bf16* kw_bf = (__bf16*)carve((size_t)512 * 768 * 2);
  __bf16* vw_bf = (__bf16*)carve((size_t)512 * 768 * 2);
  __bf16* pw_bf = (__bf16*)carve((size_t)512 * 512 * 2);
  __bf16* tx_bf = (__bf16*)carve((size_t)16 * 77 * 768 * 2);
  __bf16* klin  = (__bf16*)carve((size_t)16 * 77 * 512 * 2);
  __bf16* vlin  = (__bf16*)carve((size_t)16 * 77 * 512 * 2);
  __bf16* qT    = (__bf16*)carve((size_t)16 * 8 * 4096 * 64 * 2);
  __bf16* xnT   = (__bf16*)carve((size_t)16 * 4096 * 512 * 2);
  __bf16* oT    = xnT;   // xnT is dead after the Q GEMM -> reuse for attn out

  // weight / embedding conversions to bf16
  cvt_f32_bf16<<<(262144 + 255) / 256, 256, 0, stream>>>(qw, qw_bf, 262144);
  cvt_f32_bf16<<<(393216 + 255) / 256, 256, 0, stream>>>(kw, kw_bf, 393216);
  cvt_f32_bf16<<<(393216 + 255) / 256, 256, 0, stream>>>(vw, vw_bf, 393216);
  cvt_f32_bf16<<<(262144 + 255) / 256, 256, 0, stream>>>(pw, pw_bf, 262144);
  cvt_f32_bf16<<<(946176 + 255) / 256, 256, 0, stream>>>(text, tx_bf, 946176);

  gn_stats_k<<<512, 256, 0, stream>>>(x, stats);
  gn_apply_k<<<dim3(64, 8, 16), 256, 0, stream>>>(x, stats, gamma, beta, xnT);
  q_gemm_k<<<dim3(8, 32, 16), 256, 0, stream>>>(xnT, qw_bf, qb, qT);
  kv_gemm_k<<<dim3(5, 2, 16), 256, 0, stream>>>(tx_bf, kw_bf, vw_bf, kb, vb,
                                                klin, vlin);
  attn_k<<<dim3(16, 8, 16), 256, 0, stream>>>(qT, klin, vlin, oT);
  proj_gemm_k<<<dim3(8, 32, 16), 256, 0, stream>>>(oT, pw_bf, pb, x, out);
}
